// BaseSupertaggingModel_3642132267369
// MI455X (gfx1250) — matrix operand order repaired and checked by reading, not compiled, not run on hardware
//
#include <hip/hip_runtime.h>
#include <stdint.h>

// ---------------- problem constants ----------------
#define B_  8
#define S_  512
#define H_  768
#define L_  12
#define NH_ 12
#define DH_ 64
#define FF_ 3072
#define W_  170
#define NC_ 500
#define H1_ 1024
#define LN_EPS 1e-12f
#define MROWS (B_*S_)   // 4096

// ---------------- vector types (plain ext_vector, union-safe) ------------
typedef __attribute__((ext_vector_type(16))) __bf16 v16bf;
typedef __attribute__((ext_vector_type(8)))  float  v8f;
typedef __attribute__((ext_vector_type(4)))  unsigned int u32x4;
typedef __attribute__((ext_vector_type(2)))  unsigned int u32x2;
typedef __attribute__((ext_vector_type(4)))  float f32x4;

union FragBF {
    v16bf v;
    u32x4 q[2];
};

__device__ __forceinline__ unsigned short f32_to_bf16(float f) {
    unsigned int u = __float_as_uint(f);
    u += 0x7FFFu + ((u >> 16) & 1u);     // round-to-nearest-even
    return (unsigned short)(u >> 16);
}

__device__ __forceinline__ u32x2 pack_bf16x4(f32x4 v) {
    u32x2 r;
    r.x = (unsigned)f32_to_bf16(v.x) | ((unsigned)f32_to_bf16(v.y) << 16);
    r.y = (unsigned)f32_to_bf16(v.z) | ((unsigned)f32_to_bf16(v.w) << 16);
    return r;
}

// ================= WMMA GEMM ==============================================
// C[M,N] = act( A[M,K] * B + bias ), fp32 in/out, bf16 WMMA f32-accum.
// REQUIREMENTS (hold for every call in this model):
//   K % 64 == 0, M % 128 == 0. Only N may be ragged (N=500 head).
// transB=0: B is [K,N] row-major.  transB=1: B is [N,K] row-major (A*B^T).
// batched via blockIdx.z: z -> (zo = z/batchInner, zi = z%batchInner)
#define BM 128
#define BN 64
#define BK 64
#define BKP 72   // padded LDS row stride (ushorts): 144B rows, 16B-aligned, bank-spread

__global__ __launch_bounds__(256)
void gemm_wmma_bf16(const float* __restrict__ A, const float* __restrict__ Bg,
                    const float* __restrict__ bias, float* __restrict__ C,
                    int M, int N, int K, int lda, int ldb, int ldc,
                    int transB, int act, int batchInner,
                    long long aOut, long long aIn,
                    long long bOut, long long bIn,
                    long long cOut, long long cIn)
{
    __shared__ __align__(16) unsigned short As[2][BM * BKP];   // [m][k]
    __shared__ __align__(16) unsigned short Bs[2][BN * BKP];   // [n][k]

    const int tid  = threadIdx.x;
    const int wave = tid >> 5;          // 0..7 : m-tile index
    const int lane = tid & 31;

    const int z  = blockIdx.z;
    const int zo = z / batchInner;
    const int zi = z - zo * batchInner;
    A  += (size_t)(zo * aOut + zi * aIn);
    Bg += (size_t)(zo * bOut + zi * bIn);
    C  += (size_t)(zo * cOut + zi * cIn);

    const int blockRow = blockIdx.y * BM;
    const int blockCol = blockIdx.x * BN;
    const bool fullN   = (blockCol + BN) <= N;   // uniform: all but ragged edge

    v8f acc[4];
    acc[0] = (v8f){}; acc[1] = (v8f){}; acc[2] = (v8f){}; acc[3] = (v8f){};

    // loader coordinates (shared by A and both B paths)
    const int c4  = (tid & 15) << 2;    // 4-elem column offset 0..60
    const int r16 = tid >> 4;           // 0..15

    f32x4 stA[8];                       // staged A tile rows
    f32x4 stB[4];                       // staged B tile

    // ---- phase 1: global -> registers (issued early to hide HBM latency) ----
    auto loadTile = [&](int k0) {
        const float* Ap = A + (size_t)(blockRow + r16) * lda + (k0 + c4);
        #pragma unroll
        for (int it = 0; it < 8; ++it)
            stA[it] = *(const f32x4*)(Ap + (size_t)(it * 16) * lda);
        if (transB) {
            const float* Bp = Bg + (size_t)(blockCol + r16) * ldb + (k0 + c4);
            if (fullN) {
                #pragma unroll
                for (int it = 0; it < 4; ++it)
                    stB[it] = *(const f32x4*)(Bp + (size_t)(it * 16) * ldb);
            } else {
                #pragma unroll
                for (int it = 0; it < 4; ++it) {
                    stB[it] = (f32x4){0.f, 0.f, 0.f, 0.f};
                    if (blockCol + r16 + it * 16 < N)
                        stB[it] = *(const f32x4*)(Bp + (size_t)(it * 16) * ldb);
                }
            }
        } else {
            const int gn = blockCol + c4;    // 4 consecutive columns, coalesced
            const float* Bp = Bg + (size_t)(k0 + r16) * ldb + gn;
            if (fullN) {
                #pragma unroll
                for (int it = 0; it < 4; ++it)
                    stB[it] = *(const f32x4*)(Bp + (size_t)(it * 16) * ldb);
            } else {
                #pragma unroll
                for (int it = 0; it < 4; ++it) {
                    stB[it] = (f32x4){0.f, 0.f, 0.f, 0.f};
                    if (gn + 3 < N) {
                        stB[it] = *(const f32x4*)(Bp + (size_t)(it * 16) * ldb);
                    } else {
                        #pragma unroll
                        for (int j = 0; j < 4; ++j)
                            if (gn + j < N)
                                stB[it][j] = Bp[(size_t)(it * 16) * ldb + j];
                    }
                }
            }
        }
    };

    // ---- phase 2: convert + LDS commit (issued after the WMMA burst) ----
    auto commitTile = [&](int buf) {
        #pragma unroll
        for (int it = 0; it < 8; ++it)
            *(u32x2*)&As[buf][(r16 + it * 16) * BKP + c4] = pack_bf16x4(stA[it]);
        if (transB) {
            #pragma unroll
            for (int it = 0; it < 4; ++it)
                *(u32x2*)&Bs[buf][(r16 + it * 16) * BKP + c4] = pack_bf16x4(stB[it]);
        } else {
            #pragma unroll
            for (int it = 0; it < 4; ++it) {
                int kk = r16 + it * 16;
                Bs[buf][(c4 + 0) * BKP + kk] = f32_to_bf16(stB[it].x);
                Bs[buf][(c4 + 1) * BKP + kk] = f32_to_bf16(stB[it].y);
                Bs[buf][(c4 + 2) * BKP + kk] = f32_to_bf16(stB[it].z);
                Bs[buf][(c4 + 3) * BKP + kk] = f32_to_bf16(stB[it].w);
            }
        }
    };

    loadTile(0);
    commitTile(0);
    __syncthreads();

    const int mrow = wave * 16 + (lane & 15);
    const int kbA  = (lane < 16) ? 0 : 8;        // A frag: hi lanes start K=8
    const int kbB  = (lane < 16) ? 0 : 8;        // B frag: hi lanes start K=8
    const int ncol = lane & 15;

    int cur = 0;
    for (int k0 = 0; k0 < K; k0 += BK) {
        const bool hasNext = (k0 + BK) < K;
        if (hasNext) loadTile(k0 + BK);          // global loads in flight...

        #pragma unroll
        for (int ko = 0; ko < BK; ko += 32) {    // ...while WMMAs issue
            // load all 5 fragments into independent registers first so the
            // scheduler can use partial s_wait_dscnt between WMMAs
            FragBF a, b[4];
            a.q[0] = *(const u32x4*)&As[cur][mrow * BKP + ko + kbA];
            a.q[1] = *(const u32x4*)&As[cur][mrow * BKP + ko + kbA + 16];
            #pragma unroll
            for (int nt = 0; nt < 4; ++nt) {
                const int nrow = nt * 16 + ncol;
                b[nt].q[0] = *(const u32x4*)&Bs[cur][nrow * BKP + ko + kbB];
                b[nt].q[1] = *(const u32x4*)&Bs[cur][nrow * BKP + ko + kbB + 16];
            }
            #pragma unroll
            for (int nt = 0; nt < 4; ++nt)
                acc[nt] = __builtin_amdgcn_wmma_f32_16x16x32_bf16(
                              false, a.v, false, b[nt].v, (short)0, acc[nt],
                              false, false);
        }

        if (hasNext) commitTile(cur ^ 1);        // stores only touch other buffer
        __syncthreads();
        cur ^= 1;
    }

    // ---- epilogue: bias + activation + store (only N may be ragged) ----
    const int colInTile = lane & 15;
    const int rowOff    = (lane < 16) ? 0 : 8;
    #pragma unroll
    for (int nt = 0; nt < 4; ++nt) {
        int gcol = blockCol + nt * 16 + colInTile;
        if (!fullN && gcol >= N) continue;
        float bv = bias ? bias[gcol] : 0.f;
        float* Cp = C + (size_t)(blockRow + wave * 16 + rowOff) * ldc + gcol;
        #pragma unroll
        for (int r = 0; r < 8; ++r) {
            float v = acc[nt][r] + bv;
            if (act == 1)      v = fmaxf(v, 0.f);
            else if (act == 2) v = 0.5f * v * (1.f + erff(v * 0.70710678118f));
            Cp[(size_t)r * ldc] = v;
        }
    }
}

// ================= softmax over S=512 with mask bias =====================
__global__ __launch_bounds__(256)
void softmax_mask(float* __restrict__ scores, const int* __restrict__ mask)
{
    __shared__ float red[256];
    const int row = blockIdx.x;                  // [0, B*NH*S)
    const int b   = row / (NH_ * S_);
    float* p = scores + (size_t)row * S_;
    const int tid = threadIdx.x;
    const float scale = 0.125f;                  // 1/sqrt(64)

    float v0, v1;
    {
        int c0 = tid, c1 = tid + 256;
        float bias0 = (1.f - (float)mask[b * S_ + c0]) * -1e9f;
        float bias1 = (1.f - (float)mask[b * S_ + c1]) * -1e9f;
        v0 = p[c0] * scale + bias0;
        v1 = p[c1] * scale + bias1;
    }
    red[tid] = fmaxf(v0, v1);
    __syncthreads();
    for (int s = 128; s > 0; s >>= 1) {
        if (tid < s) red[tid] = fmaxf(red[tid], red[tid + s]);
        __syncthreads();
    }
    float m = red[0];
    __syncthreads();
    float e0 = expf(v0 - m), e1 = expf(v1 - m);
    red[tid] = e0 + e1;
    __syncthreads();
    for (int s = 128; s > 0; s >>= 1) {
        if (tid < s) red[tid] += red[tid + s];
        __syncthreads();
    }
    float inv = 1.f / red[0];
    p[tid]       = e0 * inv;
    p[tid + 256] = e1 * inv;
}

// ================= x = LayerNorm(x + t) * g + b, H=768 ===================
__global__ __launch_bounds__(256)
void add_layernorm(float* __restrict__ x, const float* __restrict__ t,
                   const float* __restrict__ g, const float* __restrict__ bta)
{
    __shared__ float red[256];
    const int row = blockIdx.x;
    const int tid = threadIdx.x;
    float* xr = x + (size_t)row * H_;
    const float* tr = t + (size_t)row * H_;

    float v[3];
    #pragma unroll
    for (int i = 0; i < 3; ++i) { int c = tid + i * 256; v[i] = xr[c] + tr[c]; }
    red[tid] = v[0] + v[1] + v[2];
    __syncthreads();
    for (int s = 128; s > 0; s >>= 1) { if (tid < s) red[tid] += red[tid + s]; __syncthreads(); }
    float mean = red[0] * (1.f / H_);
    __syncthreads();
    float d0 = v[0] - mean, d1 = v[1] - mean, d2 = v[2] - mean;
    red[tid] = d0 * d0 + d1 * d1 + d2 * d2;
    __syncthreads();
    for (int s = 128; s > 0; s >>= 1) { if (tid < s) red[tid] += red[tid + s]; __syncthreads(); }
    float inv = rsqrtf(red[0] * (1.f / H_) + LN_EPS);
    #pragma unroll
    for (int i = 0; i < 3; ++i) {
        int c = tid + i * 256;
        xr[c] = (v[i] - mean) * inv * g[c] + bta[c];
    }
}

// ================= embeddings + LayerNorm ================================
__global__ __launch_bounds__(256)
void embed_ln(const int* __restrict__ ids, const float* __restrict__ we,
              const float* __restrict__ wp, const float* __restrict__ wt,
              const float* __restrict__ g, const float* __restrict__ bta,
              float* __restrict__ x)
{
    __shared__ float red[256];
    const int row = blockIdx.x;          // b*S + s
    const int s   = row % S_;
    const int tid = threadIdx.x;
    const int id  = ids[row];

    float v[3];
    #pragma unroll
    for (int i = 0; i < 3; ++i) {
        int c = tid + i * 256;
        v[i] = we[(size_t)id * H_ + c] + wp[(size_t)s * H_ + c] + wt[c];
    }
    red[tid] = v[0] + v[1] + v[2];
    __syncthreads();
    for (int st = 128; st > 0; st >>= 1) { if (tid < st) red[tid] += red[tid + st]; __syncthreads(); }
    float mean = red[0] * (1.f / H_);
    __syncthreads();
    float d0 = v[0] - mean, d1 = v[1] - mean, d2 = v[2] - mean;
    red[tid] = d0 * d0 + d1 * d1 + d2 * d2;
    __syncthreads();
    for (int st = 128; st > 0; st >>= 1) { if (tid < st) red[tid] += red[tid + st]; __syncthreads(); }
    float inv = rsqrtf(red[0] * (1.f / H_) + LN_EPS);
    float* xr = x + (size_t)row * H_;
    #pragma unroll
    for (int i = 0; i < 3; ++i) {
        int c = tid + i * 256;
        xr[c] = (v[i] - mean) * inv * g[c] + bta[c];
    }
}

// ================= segment mean pool: f0 -> g0 ===========================
__global__ __launch_bounds__(256)
void seg_pool(const float* __restrict__ f0, const int* __restrict__ counts,
              float* __restrict__ g0)
{
    const int row = blockIdx.x;          // b*S + s
    const int b = row / S_;
    const int s = row % S_;
    const float* src = f0 + (size_t)b * S_ * H_;
    float* dst = g0 + (size_t)row * H_;
    if (s < W_) {
        int start = 0;
        for (int j = 0; j < s; ++j) start += counts[b * W_ + j];
        int cnt = counts[b * W_ + s];
        if (cnt < 1) cnt = 1;
        if (start + cnt > S_) cnt = S_ - start;
        float inv = 1.f / (float)cnt;
        for (int c = threadIdx.x; c < H_; c += 256) {
            float sum = 0.f;
            for (int p = 0; p < cnt; ++p) sum += src[(size_t)(start + p) * H_ + c];
            dst[c] = sum * inv;
        }
    } else {
        for (int c = threadIdx.x; c < H_; c += 256)
            dst[c] = src[(size_t)s * H_ + c];
    }
}

// ================= driver ================================================
extern "C" void kernel_launch(void* const* d_in, const int* in_sizes, int n_in,
                              void* d_out, int out_size, void* d_ws, size_t ws_size,
                              hipStream_t stream)
{
    (void)in_sizes; (void)n_in; (void)out_size; (void)ws_size;

    const int*   ids      = (const int*)  d_in[0];
    const int*   mask     = (const int*)  d_in[1];
    const int*   counts   = (const int*)  d_in[2];
    const float* emb_word = (const float*)d_in[3];
    const float* emb_pos  = (const float*)d_in[4];
    const float* emb_type = (const float*)d_in[5];
    const float* eln_g    = (const float*)d_in[6];
    const float* eln_b    = (const float*)d_in[7];
    const float* Wq = (const float*)d_in[8];   const float* bq = (const float*)d_in[9];
    const float* Wk = (const float*)d_in[10];  const float* bk = (const float*)d_in[11];
    const float* Wv = (const float*)d_in[12];  const float* bv = (const float*)d_in[13];
    const float* Wo = (const float*)d_in[14];  const float* bo = (const float*)d_in[15];
    const float* ln1g = (const float*)d_in[16]; const float* ln1b = (const float*)d_in[17];
    const float* Wi = (const float*)d_in[18];  const float* bi = (const float*)d_in[19];
    const float* Wf = (const float*)d_in[20];  const float* bf = (const float*)d_in[21];
    const float* ln2g = (const float*)d_in[22]; const float* ln2b = (const float*)d_in[23];
    const float* w1 = (const float*)d_in[24];  const float* b1 = (const float*)d_in[25];
    const float* w2 = (const float*)d_in[26];  const float* b2 = (const float*)d_in[27];
    float* out = (float*)d_out;

    // workspace layout (floats)
    float* ws = (float*)d_ws;
    const size_t nBSH = (size_t)MROWS * H_;
    float* x      = ws;
    float* qb     = x    + nBSH;
    float* kb     = qb   + nBSH;
    float* vb     = kb   + nBSH;
    float* ctx    = vb   + nBSH;
    float* tmp    = ctx  + nBSH;
    float* g0     = tmp  + nBSH;
    float* h1     = g0   + nBSH;                               // [M, FF]
    float* scores = h1   + (size_t)MROWS * FF_;                // [B,NH,S,S]
    float* f1     = scores + (size_t)B_ * NH_ * S_ * S_;       // [M, 1024]

    auto gemm = [&](const float* A, const float* Bm, const float* bias, float* C,
                    int M, int N, int K, int lda, int ldb, int ldc,
                    int transB, int act, int batches, int bInner,
                    long long aO, long long aI, long long bO, long long bI,
                    long long cO, long long cI) {
        dim3 grid((N + BN - 1) / BN, (M + BM - 1) / BM, batches);
        gemm_wmma_bf16<<<grid, dim3(256), 0, stream>>>(
            A, Bm, bias, C, M, N, K, lda, ldb, ldc, transB, act, bInner,
            aO, aI, bO, bI, cO, cI);
    };

    // 1) embeddings + LN
    embed_ln<<<MROWS, 256, 0, stream>>>(ids, emb_word, emb_pos, emb_type,
                                        eln_g, eln_b, x);

    const long long SH  = (long long)S_ * H_;
    const long long SS  = (long long)S_ * S_;
    const long long NSS = (long long)NH_ * SS;

    // 2) encoder layers
    for (int l = 0; l < L_; ++l) {
        const float* Wql = Wq + (size_t)l * H_ * H_;  const float* bql = bq + (size_t)l * H_;
        const float* Wkl = Wk + (size_t)l * H_ * H_;  const float* bkl = bk + (size_t)l * H_;
        const float* Wvl = Wv + (size_t)l * H_ * H_;  const float* bvl = bv + (size_t)l * H_;
        const float* Wol = Wo + (size_t)l * H_ * H_;  const float* bol = bo + (size_t)l * H_;
        const float* Wil = Wi + (size_t)l * H_ * FF_; const float* bil = bi + (size_t)l * FF_;
        const float* Wfl = Wf + (size_t)l * FF_ * H_; const float* bfl = bf + (size_t)l * H_;

        gemm(x, Wql, bql, qb, MROWS, H_, H_, H_, H_, H_, 0, 0, 1, 1, 0,0,0,0,0,0);
        gemm(x, Wkl, bkl, kb, MROWS, H_, H_, H_, H_, H_, 0, 0, 1, 1, 0,0,0,0,0,0);
        gemm(x, Wvl, bvl, vb, MROWS, H_, H_, H_, H_, H_, 0, 0, 1, 1, 0,0,0,0,0,0);

        // scores[b,h] = Q[b,:,h,:] @ K[b,:,h,:]^T   (M=S, N=S, K=DH)
        gemm(qb, kb, nullptr, scores, S_, S_, DH_, H_, H_, S_,
             /*transB=*/1, 0, B_ * NH_, NH_, SH, DH_, SH, DH_, NSS, SS);

        softmax_mask<<<B_ * NH_ * S_, 256, 0, stream>>>(scores, mask);

        // ctx[b,:,h,:] = att[b,h] @ V[b,:,h,:]      (M=S, N=DH, K=S)
        gemm(scores, vb, nullptr, ctx, S_, DH_, S_, S_, H_, H_,
             0, 0, B_ * NH_, NH_, NSS, SS, SH, DH_, SH, DH_);

        gemm(ctx, Wol, bol, tmp, MROWS, H_, H_, H_, H_, H_, 0, 0, 1, 1, 0,0,0,0,0,0);
        add_layernorm<<<MROWS, 256, 0, stream>>>(x, tmp, ln1g + (size_t)l * H_,
                                                 ln1b + (size_t)l * H_);

        gemm(x, Wil, bil, h1, MROWS, FF_, H_, H_, FF_, FF_, 0, /*gelu*/2, 1, 1, 0,0,0,0,0,0);
        gemm(h1, Wfl, bfl, tmp, MROWS, H_, FF_, FF_, H_, H_, 0, 0, 1, 1, 0,0,0,0,0,0);
        add_layernorm<<<MROWS, 256, 0, stream>>>(x, tmp, ln2g + (size_t)l * H_,
                                                 ln2b + (size_t)l * H_);
    }

    // 3) segment mean pool
    seg_pool<<<MROWS, 256, 0, stream>>>(x, counts, g0);

    // 4) heads
    gemm(g0, w1, b1, f1, MROWS, H1_, H_, H_, H1_, H1_, 0, /*relu*/1, 1, 1, 0,0,0,0,0,0);
    gemm(f1, w2, b2, out, MROWS, NC_, H1_, H1_, NC_, NC_, 0, /*relu*/1, 1, 1, 0,0,0,0,0,0);
}
